// BinaryTreeLSTM_34084860461652
// MI455X (gfx1250) — compile-verified, matrix operand order checked by reading, not run on hardware
//
#include <hip/hip_runtime.h>
#include <hip/hip_bf16.h>

// ---------------- CDNA5 WMMA types ----------------
typedef __attribute__((ext_vector_type(16))) __bf16 v16bf;
typedef __attribute__((ext_vector_type(8)))  float  v8f;
typedef int v4i_vs __attribute__((vector_size(16)));   // matches builtin's int4 pointee

#define B_        256
#define NN        1023          // nodes per tree
#define HH        128
#define XX        128
#define LL        8
#define DD        9
#define LEAVES    512
#define ROWS_LEAF (B_ * LEAVES) // 131072

// round-to-nearest-even f32 -> bf16 without relying on __bf16 conversions
static __device__ __forceinline__ __bf16 tobf(float x) {
    unsigned int u = __float_as_uint(x);
    unsigned int r = (u + 0x7FFFu + ((u >> 16) & 1u)) >> 16;
    unsigned short s = (unsigned short)r;
    return __builtin_bit_cast(__bf16, s);
}
static __device__ __forceinline__ float sigm(float x) { return 1.f / (1.f + __expf(-x)); }

union AFrag { v16bf v; unsigned int u[8]; };

// k offset of VGPR pair v for this lane-half per ISA 16-bit A layout (CDNA5 ISA)
static __device__ __forceinline__ int kbase(int v, int half) {
    return (v < 4 ? 2 * v : 16 + 2 * (v - 4)) + 8 * half;
}

// ---- async global->LDS copy (GLOBAL_LOAD_ASYNC_TO_LDS_B128, ASYNCcnt) with fallback ----
#if defined(__has_builtin)
#  if __has_builtin(__builtin_amdgcn_global_load_async_to_lds_b128)
#    define USE_ASYNC_LDS 1
#  endif
#endif
#ifndef USE_ASYNC_LDS
#  define USE_ASYNC_LDS 0
#endif

static __device__ __forceinline__ void cp16(void* dst_lds, const void* src_g) {
#if USE_ASYNC_LDS
    __builtin_amdgcn_global_load_async_to_lds_b128(
        (__attribute__((address_space(1))) v4i_vs*)src_g,
        (__attribute__((address_space(3))) v4i_vs*)dst_lds, 0, 0);
#else
    *(uint4*)dst_lds = *(const uint4*)src_g;
#endif
}
static __device__ __forceinline__ void cp_wait() {
#if USE_ASYNC_LDS
    asm volatile("s_wait_asynccnt 0x0" ::: "memory");
#endif
}

// ---------------- Kernel 1: leaf embedding masked mean -> bf16 ----------------
__global__ __launch_bounds__(256) void k_embed_mean(
    const int* __restrict__ tok, const float* __restrict__ mask,
    const float* __restrict__ emb, __bf16* __restrict__ mean)
{
    int row = blockIdx.x * 2 + (threadIdx.x >> 7);   // compact leaf row [0, ROWS_LEAF)
    int dim = threadIdx.x & 127;
    int b = row >> 9, j = row & 511;
    size_t g = (size_t)b * NN + LEAVES - 1 + j;      // global (B*N) row of this leaf
    float acc = 0.f, ms = 0.f;
#pragma unroll
    for (int t = 0; t < LL; ++t) {
        int   tk = tok[g * LL + t];
        float mk = mask[g * LL + t];
        acc += emb[(size_t)tk * XX + dim] * mk;
        ms  += mk;
    }
    float v = (ms != 0.f) ? acc / ms : 0.f;          // jnp.nan_to_num
    mean[(size_t)row * XX + dim] = tobf(v);
}

// ---------------- Kernel 2: pack [srcA | srcB] (K x N) f32 -> bf16 WMMA-B fragments ----
// dst layout: (((kt*NT + nt)*32 + lane)*16 + e), one contiguous 32B fragment per lane.
__global__ __launch_bounds__(256) void k_pack_b(
    const float* __restrict__ srcA, int ldA, int nA,
    const float* __restrict__ srcB, int ldB,
    __bf16* __restrict__ dst, int KT, int NT)
{
    int p = blockIdx.x * blockDim.x + threadIdx.x;
    int total = KT * NT * 512;
    if (p >= total) return;
    int e    = p & 15;
    int lane = (p >> 4) & 31;
    int rem  = p >> 9;
    int nt   = rem % NT;
    int kt   = rem / NT;
    int v = e >> 1, half = lane >> 4;
    int k = kt * 32 + kbase(v, half) + (e & 1);
    int n = nt * 16 + (lane & 15);
    float val = (n < nA) ? srcA[(size_t)k * ldA + n]
                         : srcB[(size_t)k * ldB + (n - nA)];
    dst[p] = tobf(val);
}

// ---------------- Kernel 3: leaf GEMM (mean @ W_iou) + leaf LSTM cell ----------------
// 32-row M tile: two A fragments share each B fragment -> 24 WMMA/wave.
__global__ __launch_bounds__(256) void k_leaf(
    const __bf16* __restrict__ mean, const __bf16* __restrict__ wpack,
    const float* __restrict__ b_iou, const float* __restrict__ c0,
    __bf16* __restrict__ h_bf, float* __restrict__ c_ws)
{
    // aliased: A tile (32x128 bf16 = 8 KB) during GEMM; iou tile (16x384 f32 = 24 KB) per phase
    __shared__ __align__(16) char smem[16 * 384 * 4];
    __bf16* At  = (__bf16*)smem;
    float*  iou = (float*)smem;
    int tid = threadIdx.x;
    int rowBase = blockIdx.x * 32;

    { // A tile: 32 contiguous mean rows = 8 KB; thread copies 32 B
        int r = tid >> 3, seg = tid & 7;
        const uint4* src = (const uint4*)(mean + (size_t)(rowBase + r) * XX) + seg * 2;
        uint4* dst = (uint4*)(At + r * 128) + seg * 2;
        cp16(dst + 0, src + 0);
        cp16(dst + 1, src + 1);
    }
    cp_wait();
    __syncthreads();

    int wave = tid >> 5, lane = tid & 31, half = lane >> 4, mr = lane & 15;
    const v16bf* bp = (const v16bf*)wpack;
    v8f acc[2][3] = {};
#pragma unroll
    for (int kt = 0; kt < 4; ++kt) {
        AFrag af0, af1;
#pragma unroll
        for (int v = 0; v < 8; ++v) {
            int ko = kt * 32 + kbase(v, half);
            af0.u[v] = *(const unsigned int*)&At[mr * 128 + ko];
            af1.u[v] = *(const unsigned int*)&At[(16 + mr) * 128 + ko];
        }
#pragma unroll
        for (int j2 = 0; j2 < 3; ++j2) {
            int nt = wave * 3 + j2;                     // 8 waves x 3 = 24 n-tiles
            v16bf bf = bp[(kt * 24 + nt) * 32 + lane];  // one 32B fragment load
            acc[0][j2] = __builtin_amdgcn_wmma_f32_16x16x32_bf16(
                false, af0.v, false, bf, (short)0, acc[0][j2], false, false);
            acc[1][j2] = __builtin_amdgcn_wmma_f32_16x16x32_bf16(
                false, af1.v, false, bf, (short)0, acc[1][j2], false, false);
        }
    }
    __syncthreads();   // A-tile lifetime ends; smem becomes the iou tile

#pragma unroll
    for (int ph = 0; ph < 2; ++ph) {
#pragma unroll
        for (int j2 = 0; j2 < 3; ++j2)
#pragma unroll
            for (int r = 0; r < 8; ++r)
                iou[(r + 8 * half) * 384 + (wave * 3 + j2) * 16 + mr] = acc[ph][j2][r];
        __syncthreads();
        // epilogue: 16 rows x 128 dims, 8 elems/thread
#pragma unroll
        for (int it = 0; it < 8; ++it) {
            int idx = it * 256 + tid;
            int m = idx >> 7, hd = idx & 127;
            int lrow = rowBase + ph * 16 + m;
            int b = lrow >> 9, j = lrow & 511;
            size_t g = ((size_t)b * NN + LEAVES - 1 + j) * HH + hd;
            float i_ = iou[m * 384 + hd]        + b_iou[hd];
            float o_ = iou[m * 384 + 128 + hd]  + b_iou[128 + hd];
            float u_ = iou[m * 384 + 256 + hd]  + b_iou[256 + hd];
            float cn = sigm(i_) * tanhf(u_) + c0[g];
            float hn = sigm(o_) * tanhf(cn);
            c_ws[g] = cn;
            h_bf[g] = tobf(hn);
        }
        __syncthreads();
    }
}

// ---------------- Kernel 4: one tree level: h_cat @ [Uf_W|U_iou] + cell ----------------
// 32-row M tile -> 80 WMMA/wave, B fragments loaded once per (kt, nt).
__global__ __launch_bounds__(256) void k_level(
    const __bf16* __restrict__ h_bf, const float* __restrict__ c_ws_in,
    const __bf16* __restrict__ wcat, const float* __restrict__ b_iou,
    const float* __restrict__ Uf_b, __bf16* __restrict__ h_out_bf,
    float* __restrict__ c_ws_out, float* __restrict__ out, int d)
{
    // aliased: A tile (32x256 bf16 = 16 KB) during GEMM; acc tile (16x640 f32 = 40 KB) per phase
    __shared__ __align__(16) char smem[16 * 640 * 4];
    __bf16* At   = (__bf16*)smem;
    float*  accs = (float*)smem;
    int tid = threadIdx.x;
    int rowsTile = blockIdx.x * 32;
    int levelBase = (1 << d) - 1;

    { // A tile: children 2n+1,2n+2 adjacent -> contiguous 512 B per h_cat row; thread copies 64 B
        int r = tid >> 3, seg = tid & 7;
        int lrow = rowsTile + r;
        int bb = lrow >> d, j = lrow - (bb << d);
        int node = levelBase + j;
        size_t gl = ((size_t)bb * NN + 2 * node + 1) * HH;  // left child; right = +128
        const uint4* src = (const uint4*)(h_bf + gl) + seg * 4;
        uint4* dst = (uint4*)(At + r * 256) + seg * 4;
        cp16(dst + 0, src + 0);
        cp16(dst + 1, src + 1);
        cp16(dst + 2, src + 2);
        cp16(dst + 3, src + 3);
    }
    cp_wait();
    __syncthreads();

    int wave = tid >> 5, lane = tid & 31, half = lane >> 4, mr = lane & 15;
    const v16bf* bp = (const v16bf*)wcat;
    v8f acc[2][5] = {};
#pragma unroll
    for (int kt = 0; kt < 8; ++kt) {
        if (kt < 7)  // keep next k-tile's B fragments flowing through L0/L2
            __builtin_prefetch(&bp[((kt + 1) * 40 + wave * 5) * 32 + lane], 0, 3);
        AFrag af0, af1;
#pragma unroll
        for (int v = 0; v < 8; ++v) {
            int ko = kt * 32 + kbase(v, half);
            af0.u[v] = *(const unsigned int*)&At[mr * 256 + ko];
            af1.u[v] = *(const unsigned int*)&At[(16 + mr) * 256 + ko];
        }
#pragma unroll
        for (int j2 = 0; j2 < 5; ++j2) {
            int nt = wave * 5 + j2;                     // 8 waves x 5 = 40 n-tiles
            v16bf bf = bp[(kt * 40 + nt) * 32 + lane];
            acc[0][j2] = __builtin_amdgcn_wmma_f32_16x16x32_bf16(
                false, af0.v, false, bf, (short)0, acc[0][j2], false, false);
            acc[1][j2] = __builtin_amdgcn_wmma_f32_16x16x32_bf16(
                false, af1.v, false, bf, (short)0, acc[1][j2], false, false);
        }
    }
    __syncthreads();   // A-tile lifetime ends; smem becomes the acc tile

#pragma unroll
    for (int ph = 0; ph < 2; ++ph) {
#pragma unroll
        for (int j2 = 0; j2 < 5; ++j2)
#pragma unroll
            for (int r = 0; r < 8; ++r)
                accs[(r + 8 * half) * 640 + (wave * 5 + j2) * 16 + mr] = acc[ph][j2][r];
        __syncthreads();
        // fused cell epilogue: cols [0,256)=f-gates pre-act, [256,640)=iou
#pragma unroll
        for (int it = 0; it < 8; ++it) {
            int idx = it * 256 + tid;
            int m = idx >> 7, hd = idx & 127;
            int lrow = rowsTile + ph * 16 + m;
            int bb = lrow >> d, j = lrow - (bb << d);
            int node = levelBase + j;
            size_t glm = ((size_t)bb * NN + 2 * node + 1) * HH;
            float fl = sigm(accs[m * 640 + hd]       + Uf_b[hd]);
            float fr = sigm(accs[m * 640 + 128 + hd] + Uf_b[128 + hd]);
            float cin = fl * c_ws_in[glm + hd] + fr * c_ws_in[glm + 128 + hd];
            float i_ = accs[m * 640 + 256 + hd] + b_iou[hd];
            float o_ = accs[m * 640 + 384 + hd] + b_iou[128 + hd];
            float u_ = accs[m * 640 + 512 + hd] + b_iou[256 + hd];
            float cn = sigm(i_) * tanhf(u_) + cin;
            float hn = sigm(o_) * tanhf(cn);
            size_t gn = ((size_t)bb * NN + node) * HH + hd;
            c_ws_out[gn] = cn;
            if (d == 0) out[(size_t)bb * HH + hd] = hn;   // root h -> fp32 output
            else        h_out_bf[gn] = tobf(hn);
        }
        __syncthreads();
    }
}

// ---------------- host launcher ----------------
extern "C" void kernel_launch(void* const* d_in, const int* in_sizes, int n_in,
                              void* d_out, int out_size, void* d_ws, size_t ws_size,
                              hipStream_t stream) {
    (void)in_sizes; (void)n_in; (void)out_size; (void)ws_size;
    const int*   subtok = (const int*)d_in[0];
    const float* mask   = (const float*)d_in[1];
    /* d_in[2] = h (zeros, never needed: every consumed h is computed) */
    const float* c0     = (const float*)d_in[3];
    const float* emb    = (const float*)d_in[4];
    const float* W_iou  = (const float*)d_in[5];
    const float* U_iou  = (const float*)d_in[6];
    const float* b_iou  = (const float*)d_in[7];
    const float* Uf_W   = (const float*)d_in[8];
    const float* Uf_b   = (const float*)d_in[9];
    float* out = (float*)d_out;

    char* ws = (char*)d_ws;
    size_t off = 0;
    auto carve = [&](size_t bytes) { size_t o = off; off += (bytes + 255) & ~(size_t)255; return o; };
    __bf16* mean_bf   = (__bf16*)(ws + carve((size_t)ROWS_LEAF * XX * 2));      // 33.5 MB
    __bf16* h_bf      = (__bf16*)(ws + carve((size_t)B_ * NN * HH * 2));        // 67 MB
    float*  c_ws      = (float*) (ws + carve((size_t)B_ * NN * HH * 4));        // 134 MB
    __bf16* wiou_pack = (__bf16*)(ws + carve((size_t)4 * 24 * 512 * 2));        // 96 KB
    __bf16* wcat_pack = (__bf16*)(ws + carve((size_t)8 * 40 * 512 * 2));        // 320 KB

    // 1) leaf embedding masked-mean (only leaf rows are ever consumed)
    k_embed_mean<<<ROWS_LEAF / 2, 256, 0, stream>>>(subtok, mask, emb, mean_bf);

    // 2) weight packs (fp32 -> bf16 WMMA-B fragment layout)
    k_pack_b<<<(4 * 24 * 512 + 255) / 256, 256, 0, stream>>>(
        W_iou, 384, 384, W_iou, 384, wiou_pack, 4, 24);
    k_pack_b<<<(8 * 40 * 512 + 255) / 256, 256, 0, stream>>>(
        Uf_W, 256, 256, U_iou, 384, wcat_pack, 8, 40);

    // 3) leaf GEMM + leaf cell (32-row tiles)
    k_leaf<<<ROWS_LEAF / 32, 256, 0, stream>>>(mean_bf, wiou_pack, b_iou, c0, h_bf, c_ws);

    // 4) bottom-up tree levels d = 8..0 (fused [Uf_W | U_iou] GEMM + cell)
    for (int d = DD - 1; d >= 0; --d) {
        int blocks = (B_ << d) / 32;
        k_level<<<blocks, 256, 0, stream>>>(h_bf, c_ws, wcat_pack, b_iou, Uf_b,
                                            h_bf, c_ws, out, d);
    }
}
